// STEAM_spatial_attention_86105504350910
// MI455X (gfx1250) — matrix operand order repaired and checked by reading, not compiled, not run on hardware
//
#include <hip/hip_runtime.h>
#include <hip/hip_bf16.h>
#include <math.h>

// ---------------- problem constants ----------------
#define BB 16
#define NN 256
#define TT 176
#define SS 64
#define HH 128
#define TOPK 15
#define SUB 32            // subspace size for eigensolver (>= TOPK, 2 WMMA N-tiles)
#define POWER_ITERS 20

typedef __attribute__((ext_vector_type(16))) _Float16 v16h;
typedef __attribute__((ext_vector_type(8)))  _Float16 v8h;
typedef __attribute__((ext_vector_type(4)))  _Float16 v4h;
typedef __attribute__((ext_vector_type(8)))  float    v8f;

// ---------------- workspace layout (float offsets) ----------------
#define OFS_Y        0            // 4096*64
#define OFS_XE       262144       // 4096*64
#define OFS_BN1SUM   524288       // 64
#define OFS_BN1SQ    524352       // 64
#define OFS_BN2SUM   524416       // 128
#define OFS_BN2SQ    524544       // 128
#define OFS_BN1SC    524672       // 64
#define OFS_BN1SH    524736       // 64
#define OFS_BN2SC    524800      // 128
#define OFS_BN2SH    524928      // 128
#define OFS_SIM      525056      // 16*256*256
#define OFS_GATE     1573632     // 16*256*15
#define STAT_ZERO_N  384         // bn1sum..bn2sq contiguous

// ---------------- WMMA operand loaders (vectorized, CDNA5 wave32 layouts) ----------------
// A (16x32 f16): lane holds row m=lane&15; halves p: K = hi*16 + p
//   -> two contiguous 8-half (16B) runs per lane: [hi*16, +7] and [hi*16+8, +7].
__device__ __forceinline__ v16h load_A_packed(const _Float16* rowbase, int hi) {
  v8h lo = *(const v8h*)(rowbase + (hi << 4));
  v8h hh = *(const v8h*)(rowbase + (hi << 4) + 8);
  v16h a;
#pragma unroll
  for (int p = 0; p < 8; ++p) { a[p] = lo[p]; a[p + 8] = hh[p]; }
  return a;
}

// A from global f32 rows via float4 loads (16B aligned); tail = second 16-K run OOB -> 0
__device__ __forceinline__ v16h load_A_gf32(const float* rowbase, int hi, bool tail) {
  v16h a;
  const float4* q0 = (const float4*)(rowbase + (hi << 3));
  float4 f0 = q0[0], f1 = q0[1];
  a[0] = (_Float16)f0.x; a[1] = (_Float16)f0.y; a[2] = (_Float16)f0.z; a[3] = (_Float16)f0.w;
  a[4] = (_Float16)f1.x; a[5] = (_Float16)f1.y; a[6] = (_Float16)f1.z; a[7] = (_Float16)f1.w;
  if (!tail) {
    const float4* q1 = (const float4*)(rowbase + 16 + (hi << 3));
    float4 g0 = q1[0], g1 = q1[1];
    a[8]  = (_Float16)g0.x; a[9]  = (_Float16)g0.y; a[10] = (_Float16)g0.z; a[11] = (_Float16)g0.w;
    a[12] = (_Float16)g1.x; a[13] = (_Float16)g1.y; a[14] = (_Float16)g1.z; a[15] = (_Float16)g1.w;
  } else {
#pragma unroll
    for (int p = 8; p < 16; ++p) a[p] = (_Float16)0.f;
  }
  return a;
}

// B (32x16 f16, KxN): lane holds N=lane&15, K = hi*16 + p. Source stored N-major
// (row = N, col = K) makes the lane's 16 halves one contiguous 32B load.
__device__ __forceinline__ v16h load_B_packed(const _Float16* p) {
  return *(const v16h*)p;
}

// ---------------- small utility kernels ----------------
__global__ void k_zero(float* p, int n) {
  int i = blockIdx.x * blockDim.x + threadIdx.x;
  if (i < n) p[i] = 0.f;
}

__global__ void k_bn_finalize(const float* __restrict__ sum, const float* __restrict__ sq,
                              const float* __restrict__ g, const float* __restrict__ bta,
                              float* __restrict__ scale, float* __restrict__ shift, float count) {
  int c = threadIdx.x;
  float mean = sum[c] / count;
  float var  = sq[c] / count - mean * mean;
  float rstd = rsqrtf(var + 1e-5f);
  float sc   = g[c] * rstd;
  scale[c] = sc;
  shift[c] = bta[c] - mean * sc;
}

__global__ void k_apply_bn1(const float* __restrict__ y, const float* __restrict__ scale,
                            const float* __restrict__ shift, float* __restrict__ xe) {
  int idx = blockIdx.x * 256 + threadIdx.x;   // 262144 total
  int s = idx & 63;
  xe[idx] = fmaxf(0.f, y[idx] * scale[s] + shift[s]);
}

// ---------------- kernel 1: embedding GEMM (X @ conv_w^T) + BN1 stats ----------------
// M=4096 rows, K=176 (padded 192), N=64. 32 blocks x 256 thr (8 waves); wave = one 16-row tile.
__global__ __launch_bounds__(256) void k_embed(const float* __restrict__ X,
                                               const float* __restrict__ conv_w,
                                               const float* __restrict__ conv_b,
                                               float* __restrict__ y,
                                               float* __restrict__ gsum, float* __restrict__ gsq) {
  __shared__ __align__(32) _Float16 cw[64][192];   // N-major: [s][t], zero-padded K
  __shared__ float s1l[64], s2l[64];
  int tid = threadIdx.x;
  for (int idx = tid; idx < 64 * 192; idx += 256) {
    int s = idx / 192, t = idx % 192;
    cw[s][t] = (t < TT) ? (_Float16)conv_w[s * TT + t] : (_Float16)0.f;
  }
  if (tid < 64) { s1l[tid] = 0.f; s2l[tid] = 0.f; }
  __syncthreads();

  int lane = tid & 31, wv = tid >> 5;
  int n = lane & 15, hi = lane >> 4;
  int mbase = blockIdx.x * 128 + wv * 16;
  const float* xrow = X + (mbase + n) * TT;   // lane's A row

  // hoist all 6 A K-chunks into VGPRs (reused across the 4 N-tiles)
  v16h at[6];
#pragma unroll
  for (int kc = 0; kc < 6; ++kc) at[kc] = load_A_gf32(xrow + kc * 32, hi, kc == 5);

  for (int nt = 0; nt < 4; ++nt) {
    v8f c = {};
#pragma unroll
    for (int kc = 0; kc < 6; ++kc) {
      v8h bl = *(const v8h*)&cw[nt * 16 + n][kc * 32 + hi * 8];
      v8h bh = *(const v8h*)&cw[nt * 16 + n][kc * 32 + 16 + hi * 8];
      v16h bm = __builtin_shufflevector(bl, bh, 0, 1, 2, 3, 4, 5, 6, 7, 8, 9, 10, 11, 12, 13, 14, 15);
      c = __builtin_amdgcn_wmma_f32_16x16x32_f16(false, at[kc], false, bm, (short)0, c, false, false);
    }
    int s = nt * 16 + n;
    float bias = conv_b[s];
    float p = 0.f, p2 = 0.f;
#pragma unroll
    for (int r = 0; r < 8; ++r) {
      float v = c[r] + bias;
      y[(mbase + r + hi * 8) * SS + s] = v;
      p += v; p2 += v * v;
    }
    atomicAdd(&s1l[s], p);
    atomicAdd(&s2l[s], p2);
  }
  __syncthreads();
  if (tid < 64) { atomicAdd(&gsum[tid], s1l[tid]); atomicAdd(&gsq[tid], s2l[tid]); }
}

// ---------------- shared piece: build dist rows in LDS (f16, vectorized) ----------------
__device__ __forceinline__ void build_dist(_Float16 (*dh)[64], const float* __restrict__ xe,
                                           const float* __restrict__ xi, int bi, int tid) {
  const float* xj = xe + (bi * NN + tid) * SS;
#pragma unroll
  for (int s0 = 0; s0 < 64; s0 += 8) {
    float4 a0 = *(const float4*)(xj + s0);
    float4 a1 = *(const float4*)(xj + s0 + 4);
    v8h d8;
    float df;
    df = xi[s0 + 0] - a0.x; d8[0] = (_Float16)sqrtf(df * df + 1e-9f);
    df = xi[s0 + 1] - a0.y; d8[1] = (_Float16)sqrtf(df * df + 1e-9f);
    df = xi[s0 + 2] - a0.z; d8[2] = (_Float16)sqrtf(df * df + 1e-9f);
    df = xi[s0 + 3] - a0.w; d8[3] = (_Float16)sqrtf(df * df + 1e-9f);
    df = xi[s0 + 4] - a1.x; d8[4] = (_Float16)sqrtf(df * df + 1e-9f);
    df = xi[s0 + 5] - a1.y; d8[5] = (_Float16)sqrtf(df * df + 1e-9f);
    df = xi[s0 + 6] - a1.z; d8[6] = (_Float16)sqrtf(df * df + 1e-9f);
    df = xi[s0 + 7] - a1.w; d8[7] = (_Float16)sqrtf(df * df + 1e-9f);
    *(v8h*)&dh[tid][s0] = d8;
  }
}

// ---------------- kernel 2: fused dist + Linear1, pass 1 (BN2 stats only) ----------------
__global__ __launch_bounds__(256) void k_pass1(const float* __restrict__ xe,
                                               const float* __restrict__ w1,
                                               const float* __restrict__ b1,
                                               float* __restrict__ gsum, float* __restrict__ gsq) {
  __shared__ __align__(32) _Float16 w1t[128][64];  // N-major: [f][s] (natural w1 layout)
  __shared__ __align__(32) _Float16 dh[256][64];
  __shared__ float xi[64];
  __shared__ float ls[128], ls2[128];
  int tid = threadIdx.x;
  int bi = blockIdx.x >> 8, ii = blockIdx.x & 255;

  for (int idx = tid; idx < 128 * 64 / 8; idx += 256) {   // v8h staging of w1 (f16)
    int f = (idx * 8) >> 6, s0 = (idx * 8) & 63;
    float4 a0 = *(const float4*)(w1 + f * SS + s0);
    float4 a1 = *(const float4*)(w1 + f * SS + s0 + 4);
    v8h h8;
    h8[0] = (_Float16)a0.x; h8[1] = (_Float16)a0.y; h8[2] = (_Float16)a0.z; h8[3] = (_Float16)a0.w;
    h8[4] = (_Float16)a1.x; h8[5] = (_Float16)a1.y; h8[6] = (_Float16)a1.z; h8[7] = (_Float16)a1.w;
    *(v8h*)&w1t[f][s0] = h8;
  }
  if (tid < 64)  xi[tid] = xe[(bi * NN + ii) * SS + tid];
  if (tid < 128) { ls[tid] = 0.f; ls2[tid] = 0.f; }
  __syncthreads();

  build_dist(dh, xe, xi, bi, tid);
  __syncthreads();

  int lane = tid & 31, wv = tid >> 5;
  int n = lane & 15, hi = lane >> 4;
  for (int mi = 0; mi < 2; ++mi) {
    int mt = 2 * wv + mi;
    const _Float16* arow = &dh[mt * 16 + n][0];
    v16h a0 = load_A_packed(arow, hi);
    v16h a1 = load_A_packed(arow + 32, hi);
    for (int nt = 0; nt < 8; ++nt) {
      const _Float16* brow = &w1t[nt * 16 + n][0];
      v8f c = {};
      c = __builtin_amdgcn_wmma_f32_16x16x32_f16(false, a0, false,
            load_B_packed(brow + hi * 16), (short)0, c, false, false);
      c = __builtin_amdgcn_wmma_f32_16x16x32_f16(false, a1, false,
            load_B_packed(brow + 32 + hi * 16), (short)0, c, false, false);
      int f = nt * 16 + n;
      float bf = b1[f];
      float p = 0.f, p2 = 0.f;
#pragma unroll
      for (int r = 0; r < 8; ++r) { float h = c[r] + bf; p += h; p2 += h * h; }
      atomicAdd(&ls[f], p);
      atomicAdd(&ls2[f], p2);
    }
  }
  __syncthreads();
  if (tid < 128) { atomicAdd(&gsum[tid], ls[tid]); atomicAdd(&gsq[tid], ls2[tid]); }
}

// ---------------- kernel 3: pass 2 — recompute, BN2+ReLU+Linear2+sigmoid -> Sim ----------------
__global__ __launch_bounds__(256) void k_pass2(const float* __restrict__ xe,
                                               const float* __restrict__ w1,
                                               const float* __restrict__ b1,
                                               const float* __restrict__ sc2,
                                               const float* __restrict__ sh2,
                                               const float* __restrict__ w2,
                                               const float* __restrict__ b2p,
                                               float* __restrict__ sim) {
  __shared__ __align__(32) _Float16 w1t[128][64];
  __shared__ __align__(32) _Float16 dh[256][64];
  __shared__ float xi[64];
  int tid = threadIdx.x;
  int bi = blockIdx.x >> 8, ii = blockIdx.x & 255;

  for (int idx = tid; idx < 128 * 64 / 8; idx += 256) {
    int f = (idx * 8) >> 6, s0 = (idx * 8) & 63;
    float4 a0 = *(const float4*)(w1 + f * SS + s0);
    float4 a1 = *(const float4*)(w1 + f * SS + s0 + 4);
    v8h h8;
    h8[0] = (_Float16)a0.x; h8[1] = (_Float16)a0.y; h8[2] = (_Float16)a0.z; h8[3] = (_Float16)a0.w;
    h8[4] = (_Float16)a1.x; h8[5] = (_Float16)a1.y; h8[6] = (_Float16)a1.z; h8[7] = (_Float16)a1.w;
    *(v8h*)&w1t[f][s0] = h8;
  }
  if (tid < 64) xi[tid] = xe[(bi * NN + ii) * SS + tid];
  __syncthreads();

  build_dist(dh, xe, xi, bi, tid);
  __syncthreads();

  int lane = tid & 31, wv = tid >> 5;
  int n = lane & 15, hi = lane >> 4;
  float b2v = b2p[0];
  for (int mi = 0; mi < 2; ++mi) {
    int mt = 2 * wv + mi;
    const _Float16* arow = &dh[mt * 16 + n][0];
    v16h a0 = load_A_packed(arow, hi);
    v16h a1 = load_A_packed(arow + 32, hi);
    float acc[8];
#pragma unroll
    for (int r = 0; r < 8; ++r) acc[r] = 0.f;
    for (int nt = 0; nt < 8; ++nt) {
      const _Float16* brow = &w1t[nt * 16 + n][0];
      v8f c = {};
      c = __builtin_amdgcn_wmma_f32_16x16x32_f16(false, a0, false,
            load_B_packed(brow + hi * 16), (short)0, c, false, false);
      c = __builtin_amdgcn_wmma_f32_16x16x32_f16(false, a1, false,
            load_B_packed(brow + 32 + hi * 16), (short)0, c, false, false);
      int f = nt * 16 + n;
      float bf = b1[f], sc = sc2[f], sh = sh2[f], wf = w2[f];
#pragma unroll
      for (int r = 0; r < 8; ++r) {
        float h = fmaxf(0.f, (c[r] + bf) * sc + sh);
        acc[r] += h * wf;
      }
    }
#pragma unroll
    for (int r = 0; r < 8; ++r) {
      float v = acc[r];
      v += __shfl_xor(v, 8, 32);
      v += __shfl_xor(v, 4, 32);
      v += __shfl_xor(v, 2, 32);
      v += __shfl_xor(v, 1, 32);
      if (n == 0) {
        int j = mt * 16 + r + hi * 8;
        float sv = 1.f / (1.f + expf(-(v + b2v)));
        sim[(bi * NN + ii) * NN + j] = (j == ii) ? 1.f : sv;
      }
    }
  }
}

// ---------------- kernel 4: top-K eigenpairs via subspace iteration + Rayleigh-Ritz ----------------
// One workgroup (256 thr / 8 waves) per batch. S (f16) resident in LDS; V and W kept f16-
// transposed (column-major rows of 256) so S@V, W^T W and V^T W are ALL pure WMMA with
// contiguous 32B lane loads. No LDS atomics anywhere in the solver.
__device__ __forceinline__ void sv_mult(const _Float16* Sh, const _Float16* VhT,
                                        float* Wv, _Float16* WhT, int lane, int wv) {
  int n = lane & 15, hi = lane >> 4;
  for (int mi = 0; mi < 2; ++mi) {
    int mt = wv + mi * 8;                       // 16 M-tiles over 8 waves
    const _Float16* arow = Sh + (mt * 16 + n) * NN;
    v16h at[8];
#pragma unroll
    for (int kc = 0; kc < 8; ++kc) at[kc] = load_A_packed(arow + kc * 32, hi);
    for (int nt = 0; nt < 2; ++nt) {
      const _Float16* brow = VhT + (nt * 16 + n) * NN;
      v8f c = {};
#pragma unroll
      for (int kc = 0; kc < 8; ++kc) {
        v16h bm = load_B_packed(brow + kc * 32 + hi * 16);
        c = __builtin_amdgcn_wmma_f32_16x16x32_f16(false, at[kc], false, bm, (short)0, c, false, false);
      }
      int nn = nt * 16 + n;
#pragma unroll
      for (int r = 0; r < 8; ++r) {
        int row = mt * 16 + r + hi * 8;
        Wv[row * SUB + nn] = c[r];
        WhT[nn * NN + row] = (_Float16)c[r];
      }
    }
  }
}

// G(32x32) = AT^T-style product: A rows from ATmat (32x256 f16), B columns from BTmat
// (N-major 32x256 f16). Waves 0-3 each compute one 16x16 tile.
__device__ __forceinline__ void gram_wmma(const _Float16* ATmat, const _Float16* BTmat,
                                          float* G, int lane, int wv) {
  if (wv >= 4) return;
  int mt = wv >> 1, nt = wv & 1;
  int n = lane & 15, hi = lane >> 4;
  const _Float16* arow = ATmat + (mt * 16 + n) * NN;
  const _Float16* brow = BTmat + (nt * 16 + n) * NN;
  v8f c = {};
#pragma unroll
  for (int kc = 0; kc < 8; ++kc) {
    v16h a  = load_A_packed(arow + kc * 32, hi);
    v16h bm = load_B_packed(brow + kc * 32 + hi * 16);
    c = __builtin_amdgcn_wmma_f32_16x16x32_f16(false, a, false, bm, (short)0, c, false, false);
  }
#pragma unroll
  for (int r = 0; r < 8; ++r)
    G[(mt * 16 + r + hi * 8) * SUB + nt * 16 + n] = c[r];
}

__global__ __launch_bounds__(256) void k_eigen(const float* __restrict__ sim,
                                               float* __restrict__ gate) {
  extern __shared__ unsigned char smem[];
  _Float16* Sh  = (_Float16*)smem;                         // 256*256*2 = 131072
  _Float16* VhT = (_Float16*)(smem + 131072);              // 32*256*2  =  16384
  _Float16* WhT = (_Float16*)(smem + 147456);              // 32*256*2  =  16384
  float*    Wv  = (float*)(smem + 163840);                 // 256*32*4  =  32768
  float*    G   = (float*)(smem + 196608);                 // 32*32*4   =   4096
  float*    Q   = (float*)(smem + 200704);                 // 32*32*4   =   4096
  float*    ev  = (float*)(smem + 204800);                 // 32*4
  int*      ord = (int*)(smem + 204928);                   // 32*4  -> total 205056

  int tid = threadIdx.x, lane = tid & 31, wv = tid >> 5;
  int bat = blockIdx.x;
  const float* S = sim + (size_t)bat * NN * NN;

  for (int idx = tid * 4; idx < NN * NN; idx += 1024) {    // vectorized f32->f16 staging
    __builtin_prefetch(S + idx + 2048, 0, 0);              // global_prefetch_b8
    float4 f = *(const float4*)(S + idx);
    v4h h4;
    h4[0] = (_Float16)f.x; h4[1] = (_Float16)f.y; h4[2] = (_Float16)f.z; h4[3] = (_Float16)f.w;
    *(v4h*)&Sh[idx] = h4;
  }

  // deterministic pseudo-random init of V row `tid` (kept in registers)
  float vr[SUB], wr[SUB];
  for (int c = 0; c < SUB; ++c) {
    unsigned h = ((unsigned)tid * 73856093u) ^ ((unsigned)(c + 1) * 19349663u);
    h *= 2654435761u;
    vr[c] = ((float)((h >> 8) & 0xFFFF)) / 32768.f - 1.f;
    VhT[c * NN + tid] = (_Float16)vr[c];
  }
  __syncthreads();

  for (int it = 0; it < POWER_ITERS; ++it) {
    sv_mult(Sh, VhT, Wv, WhT, lane, wv);                   // W = S V  (f32 + f16^T)
    __syncthreads();
    gram_wmma(WhT, WhT, G, lane, wv);                      // G = W^T W (WMMA)
    __syncthreads();
    if (tid == 0) {  // upper Cholesky G = R^T R (in place)
      for (int k = 0; k < SUB; ++k) {
        float s = G[k * SUB + k];
        for (int m = 0; m < k; ++m) s -= G[m * SUB + k] * G[m * SUB + k];
        float rkk = sqrtf(fmaxf(s, 1e-12f));
        G[k * SUB + k] = rkk;
        float inv = 1.f / rkk;
        for (int j = k + 1; j < SUB; ++j) {
          float t = G[k * SUB + j];
          for (int m = 0; m < k; ++m) t -= G[m * SUB + k] * G[m * SUB + j];
          G[k * SUB + j] = t * inv;
        }
      }
    }
    __syncthreads();
    // per-row triangular solve: v^T R = w^T  ->  V orthonormal; refresh VhT
    for (int c = 0; c < SUB; ++c) wr[c] = Wv[tid * SUB + c];
    for (int j = 0; j < SUB; ++j) {
      float t = wr[j];
      for (int m = 0; m < j; ++m) t -= vr[m] * G[m * SUB + j];
      vr[j] = t / G[j * SUB + j];
    }
    __syncthreads();
    for (int c = 0; c < SUB; ++c) VhT[c * NN + tid] = (_Float16)vr[c];
    __syncthreads();
  }

  // Rayleigh-Ritz: W = S V, T = V^T W (both WMMA)
  sv_mult(Sh, VhT, Wv, WhT, lane, wv);
  __syncthreads();
  gram_wmma(VhT, WhT, G, lane, wv);
  __syncthreads();
  if (tid == 0) {
    // symmetrize, Q = I
    for (int r = 0; r < SUB; ++r)
      for (int c = r + 1; c < SUB; ++c) {
        float t = 0.5f * (G[r * SUB + c] + G[c * SUB + r]);
        G[r * SUB + c] = t; G[c * SUB + r] = t;
      }
    for (int idx = 0; idx < SUB * SUB; ++idx) Q[idx] = 0.f;
    for (int r = 0; r < SUB; ++r) Q[r * SUB + r] = 1.f;
    // cyclic Jacobi sweeps on 32x32
    for (int sw = 0; sw < 8; ++sw) {
      for (int p = 0; p < SUB - 1; ++p) {
        for (int q = p + 1; q < SUB; ++q) {
          float apq = G[p * SUB + q];
          if (fabsf(apq) < 1e-12f) continue;
          float app = G[p * SUB + p], aqq = G[q * SUB + q];
          float theta = 0.5f * (aqq - app) / apq;
          float t = 1.f / (fabsf(theta) + sqrtf(theta * theta + 1.f));
          if (theta < 0.f) t = -t;
          float cth = rsqrtf(t * t + 1.f), sth = t * cth;
          for (int r = 0; r < SUB; ++r) {
            float grp = G[r * SUB + p], grq = G[r * SUB + q];
            G[r * SUB + p] = cth * grp - sth * grq;
            G[r * SUB + q] = sth * grp + cth * grq;
          }
          for (int r = 0; r < SUB; ++r) {
            float gpr = G[p * SUB + r], gqr = G[q * SUB + r];
            G[p * SUB + r] = cth * gpr - sth * gqr;
            G[q * SUB + r] = sth * gpr + cth * gqr;
          }
          for (int r = 0; r < SUB; ++r) {
            float qrp = Q[r * SUB + p], qrq = Q[r * SUB + q];
            Q[r * SUB + p] = cth * qrp - sth * qrq;
            Q[r * SUB + q] = sth * qrp + cth * qrq;
          }
        }
      }
    }
    for (int c = 0; c < SUB; ++c) { ev[c] = G[c * SUB + c]; ord[c] = c; }
    // insertion sort descending by eigenvalue
    for (int a2 = 1; a2 < SUB; ++a2) {
      int o = ord[a2]; float v = ev[o];
      int m = a2 - 1;
      while (m >= 0 && ev[ord[m]] < v) { ord[m + 1] = ord[m]; --m; }
      ord[m + 1] = o;
    }
  }
  __syncthreads();
  // Ritz vectors: u_k(i) = sum_j V[i,j] Q[j, ord[k]]; gate = |u|
  for (int k = 0; k < TOPK; ++k) {
    int col = ord[k];
    float u = 0.f;
    for (int j = 0; j < SUB; ++j) u += vr[j] * Q[j * SUB + col];
    gate[(bat * NN + tid) * TOPK + k] = fabsf(u);
  }
}

// ---------------- kernel 5: gated output ----------------
__global__ void k_gating(const float* __restrict__ X, const float* __restrict__ gate,
                         float* __restrict__ out, int total) {
  int idx = blockIdx.x * blockDim.x + threadIdx.x;
  if (idx >= total) return;
  int t = idx % TT;
  int i = (idx / TT) % NN;
  int k = (idx / (TT * NN)) % TOPK;
  int b = idx / (TT * NN * TOPK);
  out[idx] = X[(b * NN + i) * TT + t] * gate[(b * NN + i) * TOPK + k];
}

// ---------------- launcher ----------------
extern "C" void kernel_launch(void* const* d_in, const int* in_sizes, int n_in,
                              void* d_out, int out_size, void* d_ws, size_t ws_size,
                              hipStream_t stream) {
  (void)in_sizes; (void)n_in; (void)out_size; (void)ws_size;
  const float* X      = (const float*)d_in[0];
  const float* conv_w = (const float*)d_in[1];
  const float* conv_b = (const float*)d_in[2];
  const float* bn1_g  = (const float*)d_in[3];
  const float* bn1_b  = (const float*)d_in[4];
  const float* w1     = (const float*)d_in[5];
  const float* b1     = (const float*)d_in[6];
  const float* bn2_g  = (const float*)d_in[7];
  const float* bn2_b  = (const float*)d_in[8];
  const float* w2     = (const float*)d_in[9];
  const float* b2     = (const float*)d_in[10];
  float* ws  = (float*)d_ws;
  float* out = (float*)d_out;

  float* y       = ws + OFS_Y;
  float* xe      = ws + OFS_XE;
  float* bn1sum  = ws + OFS_BN1SUM;
  float* bn1sq   = ws + OFS_BN1SQ;
  float* bn2sum  = ws + OFS_BN2SUM;
  float* bn2sq   = ws + OFS_BN2SQ;
  float* bn1sc   = ws + OFS_BN1SC;
  float* bn1sh   = ws + OFS_BN1SH;
  float* bn2sc   = ws + OFS_BN2SC;
  float* bn2sh   = ws + OFS_BN2SH;
  float* sim     = ws + OFS_SIM;
  float* gate    = ws + OFS_GATE;

  // zero stat accumulators (graph-capture safe, deterministic)
  k_zero<<<(STAT_ZERO_N + 255) / 256, 256, 0, stream>>>(ws + OFS_BN1SUM, STAT_ZERO_N);

  // stage 1: embedding GEMM + BN1 stats
  k_embed<<<32, 256, 0, stream>>>(X, conv_w, conv_b, y, bn1sum, bn1sq);
  k_bn_finalize<<<1, SS, 0, stream>>>(bn1sum, bn1sq, bn1_g, bn1_b, bn1sc, bn1sh,
                                      (float)(BB * NN));
  k_apply_bn1<<<(BB * NN * SS) / 256, 256, 0, stream>>>(y, bn1sc, bn1sh, xe);

  // stage 2: fused dist + MLP (two-pass for global BN2 stats)
  k_pass1<<<BB * NN, 256, 0, stream>>>(xe, w1, b1, bn2sum, bn2sq);
  k_bn_finalize<<<1, HH, 0, stream>>>(bn2sum, bn2sq, bn2_g, bn2_b, bn2sc, bn2sh,
                                      (float)(BB * NN * NN));
  k_pass2<<<BB * NN, 256, 0, stream>>>(xe, w1, b1, bn2sc, bn2sh, w2, b2, sim);

  // stage 3: top-15 eigenvectors via all-WMMA subspace iteration (205056 B dynamic LDS)
  k_eigen<<<BB, 256, 205056, stream>>>(sim, gate);

  // stage 4: gated output
  int total = BB * TOPK * NN * TT;
  k_gating<<<(total + 255) / 256, 256, 0, stream>>>(X, gate, out, total);
}
